// MultiHeadAttention_12524124635213
// MI455X (gfx1250) — matrix operand order repaired and checked by reading, not compiled, not run on hardware
//
#include <hip/hip_runtime.h>
#include <hip/hip_bf16.h>

#define DEVINL __device__ __forceinline__

constexpr int Bc = 2, Tc = 2048, Cc = 1024, Hc = 16, Dc = 64;

typedef __attribute__((ext_vector_type(16))) __bf16 v16bf;
typedef __attribute__((ext_vector_type(8)))  __bf16 v8bf;
typedef __attribute__((ext_vector_type(8)))  float  v8f;

DEVINL v16bf cat16(v8bf lo, v8bf hi) {
  return __builtin_shufflevector(lo, hi, 0,1,2,3,4,5,6,7,8,9,10,11,12,13,14,15);
}

// A-matrix fragment (16x32 bf16, MxK), contiguous k = 16*hf + s (matches ld_bfrag).
DEVINL v16bf ld_afrag(const __bf16* rowp, int k0, int hf) {
  const __bf16* p = rowp + k0 + hf * 16;
  v8bf lo = *(const v8bf*)(p);
  v8bf hi = *(const v8bf*)(p + 8);
  return cat16(lo, hi);
}

// B-matrix fragment (32x16 bf16), source stored as B^T row-major: one 32B load/lane.
DEVINL v16bf ld_bfrag(const __bf16* btrow, int k0, int hf) {
  return *(const v16bf*)(btrow + k0 + hf * 16);
}

DEVINL v8f wmma_bf16(v16bf a, v16bf b, v8f c) {
  return __builtin_amdgcn_wmma_f32_16x16x32_bf16(false, a, false, b, (short)0, c, false, false);
}

// ---------------- pack kernels ----------------
__global__ void cvt_f32_bf16(const float* __restrict__ in, __bf16* __restrict__ out, int n) {
  int i = blockIdx.x * blockDim.x + threadIdx.x;
  if (i < n) out[i] = (__bf16)in[i];
}

// (H,C,D) f32 -> (H,D,C) bf16, optionally scaled (Wq carries 1/sqrt(D)*log2(e))
__global__ void transpose_w(const float* __restrict__ in, __bf16* __restrict__ out, float scl) {
  int i = blockIdx.x * blockDim.x + threadIdx.x;   // flat over H*D*C, output-coalesced
  if (i >= Hc * Dc * Cc) return;
  int c = i % Cc;
  int d = (i / Cc) % Dc;
  int h = i / (Dc * Cc);
  out[i] = (__bf16)(in[((size_t)h * Cc + c) * Dc + d] * scl);
}

// ---------------- QKV projection ----------------
// One wave -> 32(M) x 64(N=D) tile. Ping-pong pipelined over 64-wide k-steps:
// zero register-rotation moves, loads always in flight under 8 WMMAs.
__global__ __launch_bounds__(128) void qkv_gemm(
    const __bf16* __restrict__ x,                    // (B,T,C)
    const __bf16* __restrict__ wqT,                  // (H,D,C)
    const __bf16* __restrict__ wkT,
    const __bf16* __restrict__ wvT,
    __bf16* __restrict__ q,                          // (B*H,T,D)
    __bf16* __restrict__ k,                          // (B*H,T,D)
    __bf16* __restrict__ vT) {                       // (B*H,D,T)
  const int wave = (blockIdx.x * blockDim.x + threadIdx.x) >> 5;
  const int lane = threadIdx.x & 31;
  const int lid = lane & 15, hf = lane >> 4;
  const int mtiles = Tc / 32;                        // 64
  const int per = Bc * Hc * mtiles;                  // 2048
  const int mat = wave / per;                        // 0=Q 1=K 2=V
  const int rem = wave - mat * per;
  const int bh = rem / mtiles;
  const int mt = rem - bh * mtiles;
  const int b = bh / Hc;
  const int h = bh - b * Hc;
  const int m0 = mt * 32;

  const __bf16* wT = (mat == 0) ? wqT : (mat == 1) ? wkT : wvT;
  const __bf16* xrow0 = x + ((size_t)b * Tc + m0 + lid) * Cc;
  const __bf16* xrow1 = xrow0 + (size_t)16 * Cc;
  const __bf16* wrow  = wT + ((size_t)h * Dc + lid) * Cc;   // + j*16*Cc per n-tile

  v8f acc0[4] = {}, acc1[4] = {};
  v16bf a0A = ld_afrag(xrow0, 0, hf);
  v16bf a1A = ld_afrag(xrow1, 0, hf);
  v16bf bwA[4];
#pragma unroll
  for (int j = 0; j < 4; ++j) bwA[j] = ld_bfrag(wrow + (size_t)j * 16 * Cc, 0, hf);

  for (int k0 = 0; k0 < Cc; k0 += 64) {
    // stage B at k0+32 (always in range: Cc % 64 == 0)
    v16bf a0B = ld_afrag(xrow0, k0 + 32, hf);
    v16bf a1B = ld_afrag(xrow1, k0 + 32, hf);
    v16bf bwB[4];
#pragma unroll
    for (int j = 0; j < 4; ++j) bwB[j] = ld_bfrag(wrow + (size_t)j * 16 * Cc, k0 + 32, hf);
#pragma unroll
    for (int j = 0; j < 4; ++j) acc0[j] = wmma_bf16(a0A, bwA[j], acc0[j]);
#pragma unroll
    for (int j = 0; j < 4; ++j) acc1[j] = wmma_bf16(a1A, bwA[j], acc1[j]);

    // stage A for next iteration (branch-free wrap)
    const int kA = (k0 + 64 < Cc) ? k0 + 64 : 0;
    a0A = ld_afrag(xrow0, kA, hf);
    a1A = ld_afrag(xrow1, kA, hf);
#pragma unroll
    for (int j = 0; j < 4; ++j) bwA[j] = ld_bfrag(wrow + (size_t)j * 16 * Cc, kA, hf);
#pragma unroll
    for (int j = 0; j < 4; ++j) acc0[j] = wmma_bf16(a0B, bwB[j], acc0[j]);
#pragma unroll
    for (int j = 0; j < 4; ++j) acc1[j] = wmma_bf16(a1B, bwB[j], acc1[j]);
  }

  // C/D layout: lane -> (M = r + 8*hf, N = j*16 + lid)
#pragma unroll
  for (int j = 0; j < 4; ++j) {
#pragma unroll
    for (int r = 0; r < 8; ++r) {
      const int M0 = m0 + r + hf * 8;
      const int M1 = M0 + 16;
      const int d = j * 16 + lid;
      const __bf16 v0 = (__bf16)acc0[j][r];
      const __bf16 v1 = (__bf16)acc1[j][r];
      if (mat == 2) {
        vT[((size_t)bh * Dc + d) * Tc + M0] = v0;
        vT[((size_t)bh * Dc + d) * Tc + M1] = v1;
      } else if (mat == 0) {
        q[((size_t)bh * Tc + M0) * Dc + d] = v0;
        q[((size_t)bh * Tc + M1) * Dc + d] = v1;
      } else {
        k[((size_t)bh * Tc + M0) * Dc + d] = v0;
        k[((size_t)bh * Tc + M1) * Dc + d] = v1;
      }
    }
  }
}

// ---------------- causal flash attention ----------------
// One wave -> 16 query rows. Q pre-scaled by 1/sqrt(D)*log2(e) -> softmax in exp2
// domain (bare v_exp_f32). Ping-pong K buffers; V + next-K load under softmax.
__global__ __launch_bounds__(128) void attn_fwd(
    const __bf16* __restrict__ q,    // (B*H,T,D), pre-scaled
    const __bf16* __restrict__ kk,   // (B*H,T,D)
    const __bf16* __restrict__ vT,   // (B*H,D,T)
    __bf16* __restrict__ aout) {     // (B,T,H*D)
  __shared__ __align__(16) __bf16 lds[4][16][40];    // per-wave P tile, 80B row stride
  const int wv = threadIdx.x >> 5;
  const int wave = blockIdx.x * 4 + wv;
  const int lane = threadIdx.x & 31;
  const int lid = lane & 15, hf = lane >> 4;
  const int mtiles = Tc / 16;
  const int bh = wave / mtiles;
  const int mt = wave - bh * mtiles;
  const int m0 = mt * 16;

  const __bf16* qb = q  + (size_t)bh * Tc * Dc;
  const __bf16* kb = kk + (size_t)bh * Tc * Dc;
  const __bf16* vb = vT + (size_t)bh * Dc * Tc;

  const __bf16* qrow = qb + (size_t)(m0 + lid) * Dc;
  const v16bf aq0 = ld_afrag(qrow, 0, hf);    // d 0..31
  const v16bf aq1 = ld_afrag(qrow, 32, hf);   // d 32..63

  v8f o[4] = {};
  float mrow[8], lrow[8];
#pragma unroll
  for (int r = 0; r < 8; ++r) { mrow[r] = -3.0e38f; lrow[r] = 0.0f; }

  __bf16 (*P)[40] = lds[wv];
  const int sBlocks = (m0 + 16 + 31) / 32;     // causal: keys s <= m0+15

  // preload K fragments for first block: [j*2+kc], j = s-half, kc = d-half
  v16bf bkP[4], bkQ[4];
#pragma unroll
  for (int j = 0; j < 2; ++j) {
    const __bf16* krow = kb + (size_t)(j * 16 + lid) * Dc;
    bkP[j * 2 + 0] = ld_bfrag(krow, 0, hf);
    bkP[j * 2 + 1] = ld_bfrag(krow, 32, hf);
  }

  auto step = [&](v16bf (&cur)[4], v16bf (&nxt)[4], int sb) {
    const int s0 = sb * 32;
    // scores (log2 domain, Q pre-scaled)
    v8f sc0 = {}, sc1 = {};
    sc0 = wmma_bf16(aq0, cur[0], sc0);
    sc0 = wmma_bf16(aq1, cur[1], sc0);
    sc1 = wmma_bf16(aq0, cur[2], sc1);
    sc1 = wmma_bf16(aq1, cur[3], sc1);

    // V loads for this block + K loads for next block fly under softmax
    v16bf bv[4];
#pragma unroll
    for (int jd = 0; jd < 4; ++jd)
      bv[jd] = ld_bfrag(vb + (size_t)(jd * 16 + lid) * Tc, s0, hf);
    const int sn = (sb + 1 < sBlocks) ? s0 + 32 : 0;
#pragma unroll
    for (int j = 0; j < 2; ++j) {
      const __bf16* krow = kb + (size_t)(sn + j * 16 + lid) * Dc;
      nxt[j * 2 + 0] = ld_bfrag(krow, 0, hf);
      nxt[j * 2 + 1] = ld_bfrag(krow, 32, hf);
    }

    // online softmax; row M = m0 + r + 8*hf lives in one VGPR across 16 lanes
#pragma unroll
    for (int r = 0; r < 8; ++r) {
      const int M = m0 + r + hf * 8;
      float x0 = sc0[r];
      float x1 = sc1[r];
      if (s0 + lid > M)      x0 = -3.0e38f;
      if (s0 + 16 + lid > M) x1 = -3.0e38f;
      float t = fmaxf(x0, x1);
#pragma unroll
      for (int off = 1; off < 16; off <<= 1) t = fmaxf(t, __shfl_xor(t, off, 32));
      const float mn = fmaxf(mrow[r], t);
      const float p0 = exp2f(x0 - mn);
      const float p1 = exp2f(x1 - mn);
      float rs = p0 + p1;
#pragma unroll
      for (int off = 1; off < 16; off <<= 1) rs += __shfl_xor(rs, off, 32);
      const float alpha = exp2f(mrow[r] - mn);
      lrow[r] = lrow[r] * alpha + rs;
      mrow[r] = mn;
#pragma unroll
      for (int jd = 0; jd < 4; ++jd) o[jd][r] = o[jd][r] * alpha;
      P[r + hf * 8][lid]      = (__bf16)p0;
      P[r + hf * 8][16 + lid] = (__bf16)p1;
    }
    asm volatile("s_wait_dscnt 0" ::: "memory");   // wave-local LDS RAW across lanes

    // P (16x32) as A fragment from LDS; PV with preloaded V fragments
    const v16bf ap = ld_afrag(&P[lid][0], 0, hf);
#pragma unroll
    for (int jd = 0; jd < 4; ++jd) o[jd] = wmma_bf16(ap, bv[jd], o[jd]);
  };

  int sb = 0;
  while (true) {
    step(bkP, bkQ, sb);
    if (++sb >= sBlocks) break;
    step(bkQ, bkP, sb);
    if (++sb >= sBlocks) break;
  }

  const int b = bh / Hc;
  const int h = bh - b * Hc;
#pragma unroll
  for (int r = 0; r < 8; ++r) {
    const float inv = 1.0f / lrow[r];
    const int row = m0 + r + hf * 8;
#pragma unroll
    for (int jd = 0; jd < 4; ++jd) {
      const int d = jd * 16 + lid;
      aout[((size_t)b * Tc + row) * Cc + h * Dc + d] = (__bf16)(o[jd][r] * inv);
    }
  }
}

// ---------------- output projection ----------------
// out(B*T,C) f32 = attn(B*T,C) bf16 x Wo^T. 32x64 tile/wave, ping-pong pipelined.
__global__ __launch_bounds__(128) void out_proj(
    const __bf16* __restrict__ a,      // (B*T,C)
    const __bf16* __restrict__ wo,     // (C,C) = (O,K)
    float* __restrict__ out) {         // (B*T,C) f32
  const int wave = (blockIdx.x * blockDim.x + threadIdx.x) >> 5;
  const int lane = threadIdx.x & 31;
  const int lid = lane & 15, hf = lane >> 4;
  const int ntiles = Cc / 64;          // 16
  const int mt = wave / ntiles;
  const int nt = wave - mt * ntiles;
  const int m0 = mt * 32;
  const int n0 = nt * 64;

  const __bf16* arow0 = a + (size_t)(m0 + lid) * Cc;
  const __bf16* arow1 = arow0 + (size_t)16 * Cc;
  const __bf16* wrow  = wo + (size_t)(n0 + lid) * Cc;

  v8f acc0[4] = {}, acc1[4] = {};
  v16bf a0A = ld_afrag(arow0, 0, hf);
  v16bf a1A = ld_afrag(arow1, 0, hf);
  v16bf bwA[4];
#pragma unroll
  for (int j = 0; j < 4; ++j) bwA[j] = ld_bfrag(wrow + (size_t)j * 16 * Cc, 0, hf);

  for (int k0 = 0; k0 < Cc; k0 += 64) {
    v16bf a0B = ld_afrag(arow0, k0 + 32, hf);
    v16bf a1B = ld_afrag(arow1, k0 + 32, hf);
    v16bf bwB[4];
#pragma unroll
    for (int j = 0; j < 4; ++j) bwB[j] = ld_bfrag(wrow + (size_t)j * 16 * Cc, k0 + 32, hf);
#pragma unroll
    for (int j = 0; j < 4; ++j) acc0[j] = wmma_bf16(a0A, bwA[j], acc0[j]);
#pragma unroll
    for (int j = 0; j < 4; ++j) acc1[j] = wmma_bf16(a1A, bwA[j], acc1[j]);

    const int kA = (k0 + 64 < Cc) ? k0 + 64 : 0;
    a0A = ld_afrag(arow0, kA, hf);
    a1A = ld_afrag(arow1, kA, hf);
#pragma unroll
    for (int j = 0; j < 4; ++j) bwA[j] = ld_bfrag(wrow + (size_t)j * 16 * Cc, kA, hf);
#pragma unroll
    for (int j = 0; j < 4; ++j) acc0[j] = wmma_bf16(a0B, bwB[j], acc0[j]);
#pragma unroll
    for (int j = 0; j < 4; ++j) acc1[j] = wmma_bf16(a1B, bwB[j], acc1[j]);
  }

#pragma unroll
  for (int j = 0; j < 4; ++j)
#pragma unroll
    for (int r = 0; r < 8; ++r) {
      out[(size_t)(m0 + r + hf * 8) * Cc + n0 + j * 16 + lid]      = acc0[j][r];
      out[(size_t)(m0 + 16 + r + hf * 8) * Cc + n0 + j * 16 + lid] = acc1[j][r];
    }
}

// ---------------- host ----------------
extern "C" void kernel_launch(void* const* d_in, const int* in_sizes, int n_in,
                              void* d_out, int out_size, void* d_ws, size_t ws_size,
                              hipStream_t stream) {
  const float* hx = (const float*)d_in[0];   // (B,T,C)
  const float* hwq = (const float*)d_in[1];  // (H,C,D)
  const float* hwk = (const float*)d_in[2];
  const float* hwv = (const float*)d_in[3];
  const float* hwo = (const float*)d_in[4];  // (C,C)
  float* out = (float*)d_out;

  char* w = (char*)d_ws;
  const size_t szX  = (size_t)Bc * Tc * Cc * 2;       // 8 MB
  const size_t szWh = (size_t)Hc * Dc * Cc * 2;       // 2 MB
  const size_t szWo = (size_t)Cc * Cc * 2;            // 2 MB
  const size_t szQ  = (size_t)Bc * Hc * Tc * Dc * 2;  // 8 MB

  __bf16* xbf  = (__bf16*)(w);
  __bf16* wqT  = (__bf16*)(w + szX);
  __bf16* wkT  = (__bf16*)(w + szX + szWh);
  __bf16* wvT  = (__bf16*)(w + szX + 2 * szWh);
  __bf16* wobf = (__bf16*)(w + szX + 3 * szWh);
  __bf16* qb   = (__bf16*)(w + szX + 3 * szWh + szWo);
  __bf16* kb   = (__bf16*)(w + szX + 3 * szWh + szWo + szQ);
  __bf16* vTb  = (__bf16*)(w + szX + 3 * szWh + szWo + 2 * szQ);
  __bf16* abf  = (__bf16*)(w + szX + 3 * szWh + szWo + 3 * szQ);

  const int nX = Bc * Tc * Cc;                // 4M
  cvt_f32_bf16<<<(nX + 255) / 256, 256, 0, stream>>>(hx, xbf, nX);
  const int nW = Hc * Dc * Cc;                // 1M
  const float qscale = 0.125f * 1.44269504088896341f;  // 1/sqrt(D) * log2(e)
  transpose_w<<<(nW + 255) / 256, 256, 0, stream>>>(hwq, wqT, qscale);
  transpose_w<<<(nW + 255) / 256, 256, 0, stream>>>(hwk, wkT, 1.0f);
  transpose_w<<<(nW + 255) / 256, 256, 0, stream>>>(hwv, wvT, 1.0f);
  const int nO = Cc * Cc;
  cvt_f32_bf16<<<(nO + 255) / 256, 256, 0, stream>>>(hwo, wobf, nO);

  const int qkvWaves = 3 * Bc * Hc * (Tc / 32);         // 6144
  qkv_gemm<<<qkvWaves / 4, 128, 0, stream>>>(xbf, wqT, wkT, wvT, qb, kb, vTb);

  const int attnWaves = Bc * Hc * (Tc / 16);            // 4096
  attn_fwd<<<attnWaves / 4, 128, 0, stream>>>(qb, kb, vTb, abf);

  const int projWaves = (Bc * Tc / 32) * (Cc / 64);     // 2048
  out_proj<<<projWaves / 4, 128, 0, stream>>>(abf, wobf, out);
}
